// ADGAT_7232724927263
// MI455X (gfx1250) — matrix operand order repaired and checked by reading, not compiled, not run on hardware
//
#include <hip/hip_runtime.h>
#include <hip/hip_bf16.h>

#define DEV __device__ __forceinline__

typedef __attribute__((ext_vector_type(16))) __bf16 bf16x16;
typedef __attribute__((ext_vector_type(8)))  __bf16 bf16x8;
typedef __attribute__((ext_vector_type(8)))  float  f32x8;

constexpr int F_IN  = 500;
constexpr int HEADS = 8;
constexpr int HID   = 8;
constexpr int C1    = HEADS * HID;   // 64
constexpr int NC    = 16;

DEV float lrelu02(float x) { return x > 0.f ? x : 0.2f * x; }
DEV float elu1(float x)    { return x > 0.f ? x : (__expf(x) - 1.f); }

// Monotone order-preserving float<->uint key for atomic float max.
DEV unsigned fkey(float f) {
  unsigned u = __float_as_uint(f);
  return (u & 0x80000000u) ? ~u : (u | 0x80000000u);
}
DEV float fdec(unsigned k) {
  unsigned u = (k & 0x80000000u) ? (k & 0x7FFFFFFFu) : ~k;
  return __uint_as_float(u);
}

// ---------------------------------------------------------------------------
// Generic row-major GEMM  O[n,NOUT] = A[n,KDIM] * W[KDIM,NOUT]  via bf16 WMMA.
// 256 threads = 8 waves; wave w computes rows [blk*128 + 16w, +16) x NOUT.
// Entire W staged once to LDS, TRANSPOSED [col][k] with padded stride so each
// lane's 16 B-elements are two 16B-aligned 16-byte runs, 32 bytes apart (2x ds_load_b128).
// ---------------------------------------------------------------------------
template <int KDIM, int NOUT>
__global__ __launch_bounds__(256) void k_gemm_bf16(
    const float* __restrict__ A, const float* __restrict__ W,
    float* __restrict__ O, int n)
{
  constexpr int KP  = (KDIM + 31) & ~31;
  constexpr int NT  = NOUT / 16;
  constexpr int STR = KP + 8;          // elements; mult of 8 -> 16B-aligned rows
  __shared__ __bf16 lw[NOUT * STR];    // W transposed: lw[col*STR + k]

  const int tid  = threadIdx.x;
  const int wave = tid >> 5;
  const int lane = tid & 31;
  const int l16  = lane & 15;
  const int hi   = lane >> 4;          // 0: lanes 0-15, 1: lanes 16-31

  // Stage all of W (bf16, transposed, zero-padded K) once.
  for (int i = tid; i < KP * NOUT; i += 256) {
    const int kk = i / NOUT;           // consecutive threads -> consecutive nn
    const int nn = i % NOUT;           //   => coalesced global reads of W
    lw[nn * STR + kk] = (__bf16)((kk < KDIM) ? W[kk * NOUT + nn] : 0.f);
  }
  __syncthreads();

  const int rowbase = blockIdx.x * 128 + wave * 16;
  int arow = rowbase + l16;
  if (arow >= n) arow = n - 1;         // clamp: WMMA needs full EXEC; stores guarded
  const float* __restrict__ ar = A + (long long)arow * KDIM;

  f32x8 acc[NT] = {};

  for (int kb = 0; kb < KP; kb += 32) {
    if (kb + 128 < KDIM) __builtin_prefetch(ar + kb + 128, 0, 1);

    // A fragment, 16-bit A 16x32 layout:
    // lane<16: row M=l16, K in {kb+0..7} U {kb+16..23}; lanes>=16: +8.
    bf16x16 a;
    if (kb + 32 <= KDIM) {             // full step: 4x b128 loads
      const float4* p = (const float4*)(ar + kb + hi * 8);
      const float4* q = (const float4*)(ar + kb + 16 + hi * 8);
      float4 x0 = p[0], x1 = p[1], y0 = q[0], y1 = q[1];
      a[0]  = (__bf16)x0.x; a[1]  = (__bf16)x0.y; a[2]  = (__bf16)x0.z; a[3]  = (__bf16)x0.w;
      a[4]  = (__bf16)x1.x; a[5]  = (__bf16)x1.y; a[6]  = (__bf16)x1.z; a[7]  = (__bf16)x1.w;
      a[8]  = (__bf16)y0.x; a[9]  = (__bf16)y0.y; a[10] = (__bf16)y0.z; a[11] = (__bf16)y0.w;
      a[12] = (__bf16)y1.x; a[13] = (__bf16)y1.y; a[14] = (__bf16)y1.z; a[15] = (__bf16)y1.w;
    } else {                           // tail step: guarded scalar loads
      #pragma unroll
      for (int i = 0; i < 8; ++i) {
        int k0 = kb + hi * 8 + i;
        int k1 = kb + 16 + hi * 8 + i;
        a[i]     = (__bf16)((k0 < KDIM) ? ar[k0] : 0.f);
        a[i + 8] = (__bf16)((k1 < KDIM) ? ar[k1] : 0.f);
      }
    }

    #pragma unroll
    for (int t = 0; t < NT; ++t) {
      // B fragment, 32x16: lane holds col=l16, K in {kb+hi*8+0..7} U {kb+16+hi*8+0..7}.
      const int col = t * 16 + l16;
      const bf16x8* bp = (const bf16x8*)&lw[col * STR + kb + hi * 8];
      bf16x8 b0 = bp[0], b1 = bp[2];
      bf16x16 b = __builtin_shufflevector(
          b0, b1, 0, 1, 2, 3, 4, 5, 6, 7, 8, 9, 10, 11, 12, 13, 14, 15);
      acc[t] = __builtin_amdgcn_wmma_f32_16x16x32_bf16(
          false, a, false, b, (short)0, acc[t], false, false);
    }
  }

  // C layout: VGPR j -> M = hi*8 + j, N = l16 (per 16-col tile).
  #pragma unroll
  for (int t = 0; t < NT; ++t) {
    #pragma unroll
    for (int j = 0; j < 8; ++j) {
      int r = rowbase + hi * 8 + j;
      if (r < n) O[(long long)r * NOUT + t * 16 + l16] = acc[t][j];
    }
  }
}

// ---------------------------------------------------------------------------
// Elementwise / node kernels
// ---------------------------------------------------------------------------
__global__ void k_zero(float* p, long long n) {
  long long i  = (long long)blockIdx.x * blockDim.x + threadIdx.x;
  long long st = (long long)gridDim.x * blockDim.x;
  for (; i < n; i += st) p[i] = 0.f;
}

__global__ void k_al1(const float* __restrict__ H, const float* __restrict__ as_,
                      const float* __restrict__ ad_, float* __restrict__ ALS,
                      float* __restrict__ ALD, int n) {
  int i = blockIdx.x * blockDim.x + threadIdx.x;
  if (i >= n) return;
  #pragma unroll
  for (int h = 0; h < HEADS; ++h) {
    float s = 0.f, d = 0.f;
    #pragma unroll
    for (int c = 0; c < HID; ++c) {
      float v = H[i * C1 + h * HID + c];
      s += v * as_[h * HID + c];
      d += v * ad_[h * HID + c];
    }
    ALS[i * HEADS + h] = s;
    ALD[i * HEADS + h] = d;
  }
}

__global__ void k_al2(const float* __restrict__ H, const float* __restrict__ as_,
                      const float* __restrict__ ad_, float* __restrict__ ALS,
                      float* __restrict__ ALD, int n) {
  int i = blockIdx.x * blockDim.x + threadIdx.x;
  if (i >= n) return;
  float s = 0.f, d = 0.f;
  #pragma unroll
  for (int c = 0; c < NC; ++c) {
    float v = H[i * NC + c];
    s += v * as_[c];
    d += v * ad_[c];
  }
  ALS[i] = s;
  ALD[i] = d;
}

__global__ void k_act1(const float* __restrict__ O, const float* __restrict__ b,
                       float* __restrict__ ACT, long long tot) {
  long long i  = (long long)blockIdx.x * blockDim.x + threadIdx.x;
  long long st = (long long)gridDim.x * blockDim.x;
  for (; i < tot; i += st) ACT[i] = elu1(O[i] + b[i & 63]);
}

__global__ void k_act2(const float* __restrict__ O, const float* __restrict__ b,
                       float* __restrict__ ACT, long long tot) {
  long long i  = (long long)blockIdx.x * blockDim.x + threadIdx.x;
  long long st = (long long)gridDim.x * blockDim.x;
  for (; i < tot; i += st) ACT[i] = fmaxf(O[i] + b[i & 15], 0.f);
}

__global__ void k_dinv(float* DEG, int n) {
  int i = blockIdx.x * blockDim.x + threadIdx.x;
  if (i >= n) return;
  float dg = DEG[i];
  DEG[i] = dg > 0.f ? rsqrtf(dg) : 0.f;
}

__global__ void k_bias_out(float* __restrict__ O, const float* __restrict__ bg,
                           long long tot) {
  long long i  = (long long)blockIdx.x * blockDim.x + threadIdx.x;
  long long st = (long long)gridDim.x * blockDim.x;
  for (; i < tot; i += st) O[i] += bg[i & 15];
}

// ---------------------------------------------------------------------------
// Edge kernels (self-loops appended virtually: e in [E, Etot) => s=d=e-E)
// ---------------------------------------------------------------------------
DEV void edge_sd(const long long* __restrict__ ei, long long E, long long e,
                 int& s, int& d) {
  if (e < E) { s = (int)ei[e]; d = (int)ei[E + e]; }
  else       { s = d = (int)(e - E); }
}

__global__ void k_edge_max1(const long long* __restrict__ ei, long long E,
                            long long Etot, const float* __restrict__ ALS,
                            const float* __restrict__ ALD, unsigned* __restrict__ EMAX) {
  long long e = (long long)blockIdx.x * blockDim.x + threadIdx.x;
  if (e >= Etot) return;
  int s, d; edge_sd(ei, E, e, s, d);
  #pragma unroll
  for (int h = 0; h < HEADS; ++h) {
    float ev = lrelu02(ALS[s * HEADS + h] + ALD[d * HEADS + h]);
    atomicMax(&EMAX[d * HEADS + h], fkey(ev));
  }
}

__global__ void k_edge_den1(const long long* __restrict__ ei, long long E,
                            long long Etot, const float* __restrict__ ALS,
                            const float* __restrict__ ALD,
                            const unsigned* __restrict__ EMAX,
                            float* __restrict__ DEN, float* __restrict__ DEG) {
  long long e = (long long)blockIdx.x * blockDim.x + threadIdx.x;
  if (e >= Etot) return;
  int s, d; edge_sd(ei, E, e, s, d);
  #pragma unroll
  for (int h = 0; h < HEADS; ++h) {
    float ev = lrelu02(ALS[s * HEADS + h] + ALD[d * HEADS + h]);
    float ex = __expf(ev - fdec(EMAX[d * HEADS + h]));
    atomicAdd(&DEN[d * HEADS + h], ex);
  }
  atomicAdd(&DEG[d], 1.f);   // degree for the GCN layer (same dst list)
}

__global__ void k_edge_agg1(const long long* __restrict__ ei, long long E,
                            long long Etot, const float* __restrict__ ALS,
                            const float* __restrict__ ALD,
                            const unsigned* __restrict__ EMAX,
                            const float* __restrict__ DEN,
                            const float* __restrict__ H, float* __restrict__ OUT) {
  long long e = (long long)blockIdx.x * blockDim.x + threadIdx.x;
  if (e >= Etot) return;
  int s, d; edge_sd(ei, E, e, s, d);
  #pragma unroll
  for (int h = 0; h < HEADS; ++h) {
    float ev = lrelu02(ALS[s * HEADS + h] + ALD[d * HEADS + h]);
    float ex = __expf(ev - fdec(EMAX[d * HEADS + h]));
    float al = ex / (DEN[d * HEADS + h] + 1e-16f);
    #pragma unroll
    for (int c = 0; c < HID; ++c)
      atomicAdd(&OUT[d * C1 + h * HID + c], al * H[s * C1 + h * HID + c]);
  }
}

__global__ void k_edge_max2(const long long* __restrict__ ei, long long E,
                            long long Etot, const float* __restrict__ ALS,
                            const float* __restrict__ ALD, unsigned* __restrict__ EMAX) {
  long long e = (long long)blockIdx.x * blockDim.x + threadIdx.x;
  if (e >= Etot) return;
  int s, d; edge_sd(ei, E, e, s, d);
  atomicMax(&EMAX[d], fkey(lrelu02(ALS[s] + ALD[d])));
}

__global__ void k_edge_den2(const long long* __restrict__ ei, long long E,
                            long long Etot, const float* __restrict__ ALS,
                            const float* __restrict__ ALD,
                            const unsigned* __restrict__ EMAX, float* __restrict__ DEN) {
  long long e = (long long)blockIdx.x * blockDim.x + threadIdx.x;
  if (e >= Etot) return;
  int s, d; edge_sd(ei, E, e, s, d);
  float ex = __expf(lrelu02(ALS[s] + ALD[d]) - fdec(EMAX[d]));
  atomicAdd(&DEN[d], ex);
}

__global__ void k_edge_agg2(const long long* __restrict__ ei, long long E,
                            long long Etot, const float* __restrict__ ALS,
                            const float* __restrict__ ALD,
                            const unsigned* __restrict__ EMAX,
                            const float* __restrict__ DEN,
                            const float* __restrict__ H, float* __restrict__ OUT) {
  long long e = (long long)blockIdx.x * blockDim.x + threadIdx.x;
  if (e >= Etot) return;
  int s, d; edge_sd(ei, E, e, s, d);
  float ex = __expf(lrelu02(ALS[s] + ALD[d]) - fdec(EMAX[d]));
  float al = ex / (DEN[d] + 1e-16f);
  #pragma unroll
  for (int c = 0; c < NC; ++c)
    atomicAdd(&OUT[d * NC + c], al * H[s * NC + c]);
}

__global__ void k_edge_gcn(const long long* __restrict__ ei, long long E,
                           long long Etot, const float* __restrict__ DINV,
                           const float* __restrict__ XW, float* __restrict__ OUT) {
  long long e = (long long)blockIdx.x * blockDim.x + threadIdx.x;
  if (e >= Etot) return;
  int s, d; edge_sd(ei, E, e, s, d);
  float nrm = DINV[s] * DINV[d];
  #pragma unroll
  for (int c = 0; c < NC; ++c)
    atomicAdd(&OUT[d * NC + c], nrm * XW[s * NC + c]);
}

// ---------------------------------------------------------------------------
extern "C" void kernel_launch(void* const* d_in, const int* in_sizes, int n_in,
                              void* d_out, int out_size, void* d_ws, size_t ws_size,
                              hipStream_t stream) {
  (void)n_in; (void)out_size; (void)ws_size;
  const float*     X   = (const float*)d_in[0];
  const long long* EI  = (const long long*)d_in[1];   // int64 edge_index [2,E]
  const float*     W1  = (const float*)d_in[2];
  const float*     a1s = (const float*)d_in[3];
  const float*     a1d = (const float*)d_in[4];
  const float*     b1  = (const float*)d_in[5];
  const float*     W2  = (const float*)d_in[6];
  const float*     a2s = (const float*)d_in[7];
  const float*     a2d = (const float*)d_in[8];
  const float*     b2  = (const float*)d_in[9];
  const float*     Wg  = (const float*)d_in[10];
  const float*     bg  = (const float*)d_in[11];
  float* out = (float*)d_out;

  const int       N    = in_sizes[0] / F_IN;
  const long long E    = in_sizes[1] / 2;
  const long long Etot = E + N;

  // Workspace layout (all fp32 / u32)
  float* p = (float*)d_ws;
  float* H1   = p; p += (long long)N * C1;
  float* ALS1 = p; p += (long long)N * HEADS;
  float* ALD1 = p; p += (long long)N * HEADS;
  float* ACT1 = p; p += (long long)N * C1;
  float* H2   = p; p += (long long)N * NC;
  float* ALS2 = p; p += N;
  float* ALD2 = p; p += N;
  float* ACT2 = p; p += (long long)N * NC;
  float* XW   = p; p += (long long)N * NC;
  // zeroed-every-launch accumulators (contiguous block)
  float*    zbase = p;
  unsigned* EMAX1 = (unsigned*)p; p += (long long)N * HEADS;
  float*    DEN1  = p;            p += (long long)N * HEADS;
  float*    OUT1  = p;            p += (long long)N * C1;
  unsigned* EMAX2 = (unsigned*)p; p += N;
  float*    DEN2  = p;            p += N;
  float*    OUT2  = p;            p += (long long)N * NC;
  float*    DEG   = p;            p += N;
  const long long zcount = p - zbase;

  const int gN = (N + 255) / 256;
  const int gG = (N + 127) / 128;
  const int gE = (int)((Etot + 255) / 256);

  k_zero<<<1024, 256, 0, stream>>>(zbase, zcount);
  k_zero<<<1024, 256, 0, stream>>>(out, (long long)N * NC);

  // ---- GAT layer 1 ----
  k_gemm_bf16<F_IN, C1><<<gG, 256, 0, stream>>>(X, W1, H1, N);
  k_al1<<<gN, 256, 0, stream>>>(H1, a1s, a1d, ALS1, ALD1, N);
  k_edge_max1<<<gE, 256, 0, stream>>>(EI, E, Etot, ALS1, ALD1, EMAX1);
  k_edge_den1<<<gE, 256, 0, stream>>>(EI, E, Etot, ALS1, ALD1, EMAX1, DEN1, DEG);
  k_edge_agg1<<<gE, 256, 0, stream>>>(EI, E, Etot, ALS1, ALD1, EMAX1, DEN1, H1, OUT1);
  k_act1<<<1024, 256, 0, stream>>>(OUT1, b1, ACT1, (long long)N * C1);

  // ---- GAT layer 2 ----
  k_gemm_bf16<C1, NC><<<gG, 256, 0, stream>>>(ACT1, W2, H2, N);
  k_al2<<<gN, 256, 0, stream>>>(H2, a2s, a2d, ALS2, ALD2, N);
  k_edge_max2<<<gE, 256, 0, stream>>>(EI, E, Etot, ALS2, ALD2, EMAX2);
  k_edge_den2<<<gE, 256, 0, stream>>>(EI, E, Etot, ALS2, ALD2, EMAX2, DEN2);
  k_edge_agg2<<<gE, 256, 0, stream>>>(EI, E, Etot, ALS2, ALD2, EMAX2, DEN2, H2, OUT2);
  k_act2<<<1024, 256, 0, stream>>>(OUT2, b2, ACT2, (long long)N * NC);

  // ---- GCN layer ----
  k_dinv<<<gN, 256, 0, stream>>>(DEG, N);
  k_gemm_bf16<NC, NC><<<gG, 256, 0, stream>>>(ACT2, Wg, XW, N);
  k_edge_gcn<<<gE, 256, 0, stream>>>(EI, E, Etot, DEG, XW, out);
  k_bias_out<<<1024, 256, 0, stream>>>(out, bg, (long long)N * NC);
}
